// MambaBlock_19928648253987
// MI455X (gfx1250) — matrix operand order repaired and checked by reading, not compiled, not run on hardware
//
#include <hip/hip_runtime.h>
#include <cstdint>
#include <cstddef>

// ---------------------------------------------------------------------------
// MambaBlock forward for MI455X (gfx1250, wave32, WMMA).
// GEMM1 (x @ w_in, bf16 WMMA) -> causal depthwise conv + SiLU gate
//   -> GEMM2 (act @ w_out, bf16 WMMA, fp32 out).
// Dead inputs per reference DCE: w_sel, w_dt, b_dt, A, Dp.
// Round 2: double-buffered LDS, register-staged global loads (waits land after
// the WMMA block), native __bf16 converts, B staged via column-quads so the
// transpose uses ds_store_b64 instead of 16x ds_store_b16.
// ---------------------------------------------------------------------------

typedef __bf16 bf16_t;
typedef bf16_t v16bf __attribute__((ext_vector_type(16)));
typedef bf16_t v2bf  __attribute__((ext_vector_type(2)));
typedef float  v8f   __attribute__((ext_vector_type(8)));

#define BM 128
#define BN 128
#define BK 32
#define LDS_STRIDE 40   // bf16 per LDS row (BK + 8 pad) -> 80B stride, conflict-free

#define D_MODEL 1024
#define E_DIM   2048
#define L_SEQ   4096
#define B_SZ    4

__device__ __forceinline__ unsigned short f2bf(float f) {
  bf16_t h = (bf16_t)f;                       // native cvt, RNE
  return __builtin_bit_cast(unsigned short, h);
}
__device__ __forceinline__ unsigned pack2bf(float x, float y) {
  v2bf t = {(bf16_t)x, (bf16_t)y};
  return __builtin_bit_cast(unsigned, t);
}
__device__ __forceinline__ float bf2f(unsigned short h) {
  unsigned int u = ((unsigned int)h) << 16;
  return __builtin_bit_cast(float, u);
}

struct Frag256 { uint4 lo, hi; };

// A fragment (16x32 bf16, MxK): per-lane k contiguous:
// VGPR0-3 -> k = half*16 + {0..7}, VGPR4-7 -> k = half*16 + 8 + {0..7}.
__device__ __forceinline__ v16bf load_fragA(const unsigned short* rowp, int half) {
  Frag256 t;
  t.lo = *(const uint4*)(rowp + half * 16);
  t.hi = *(const uint4*)(rowp + half * 16 + 8);
  return __builtin_bit_cast(v16bf, t);
}
// B fragment (32x16 bf16, KxN) from LDS stored transposed [n][k]:
// lanes 0-15 hold k=0..15, lanes 16-31 hold k=16..31 -> 16 contiguous bf16.
__device__ __forceinline__ v16bf load_fragB(const unsigned short* rowp, int half) {
  Frag256 t;
  const uint4* p = (const uint4*)(rowp + half * 16);
  t.lo = p[0];
  t.hi = p[1];
  return __builtin_bit_cast(v16bf, t);
}

// C = A(MxK) x B(KxN).  A_BF16: A operand already bf16 (workspace) else fp32.
// OUT_BF16: store bf16 (intermediate) else fp32 (final output).
template <bool A_BF16, bool OUT_BF16>
__global__ __launch_bounds__(256)
void gemm_bf16_wmma(const void* __restrict__ Ap, const float* __restrict__ Bp,
                    void* __restrict__ Cp, int Kd, int lda, int ldb, int ldc) {
  __shared__ __align__(16) unsigned short sA[2][BM * LDS_STRIDE];
  __shared__ __align__(16) unsigned short sB[2][BN * LDS_STRIDE];  // [n][k]

  const int tid   = threadIdx.x;
  const int lane  = tid & 31;
  const int wave  = tid >> 5;
  const int half  = lane >> 4;
  const int ln16  = lane & 15;
  const int mBase = (wave >> 2) * 64;   // 2 waves in M, 64 rows each
  const int nBase = (wave & 3) * 32;    // 4 waves in N, 32 cols each

  const size_t rowBase = (size_t)blockIdx.y * BM;
  const size_t colBase = (size_t)blockIdx.x * BN;

  // ---- register staging buffers for the next K-tile ----
  float4 aRegF[4];   // fp32 A path
  uint4  aRegH[2];   // bf16 A path
  float  bReg[16];   // B column-quads: 4 quads of 4 k-consecutive values

  auto issueA = [&](int kt) {
    if (A_BF16) {
      const unsigned short* Abf = (const unsigned short*)Ap;
#pragma unroll
      for (int i = 0; i < 2; ++i) {
        int idx = tid + i * 256;              // 512 x uint4 (8 bf16)
        int r = idx >> 2, c = (idx & 3) * 8;
        aRegH[i] = *(const uint4*)(Abf + (rowBase + r) * (size_t)lda + kt * BK + c);
      }
    } else {
      const float* Af = (const float*)Ap;
#pragma unroll
      for (int i = 0; i < 4; ++i) {
        int idx = tid + i * 256;              // 1024 x float4
        int r = idx >> 3, c = (idx & 7) * 4;
        aRegF[i] = *(const float4*)(Af + (rowBase + r) * (size_t)lda + kt * BK + c);
      }
    }
  };
  // B loaded as column quads: thread -> (n, k0..k0+3); each j-load is a fully
  // coalesced 128-float row segment across the block.
  auto issueB = [&](int kt) {
#pragma unroll
    for (int i = 0; i < 4; ++i) {
      int idx = tid + i * 256;                // 1024 quads
      int n = idx & 127;
      int k0 = (idx >> 7) * 4;
#pragma unroll
      for (int j = 0; j < 4; ++j)
        bReg[i * 4 + j] = Bp[(size_t)(kt * BK + k0 + j) * ldb + colBase + n];
    }
  };
  auto stage = [&](int buf) {
    if (A_BF16) {
#pragma unroll
      for (int i = 0; i < 2; ++i) {
        int idx = tid + i * 256;
        int r = idx >> 2, c = (idx & 3) * 8;
        *(uint4*)(&sA[buf][r * LDS_STRIDE + c]) = aRegH[i];
      }
    } else {
#pragma unroll
      for (int i = 0; i < 4; ++i) {
        int idx = tid + i * 256;
        int r = idx >> 3, c = (idx & 7) * 4;
        uint2 pk;
        pk.x = pack2bf(aRegF[i].x, aRegF[i].y);
        pk.y = pack2bf(aRegF[i].z, aRegF[i].w);
        *(uint2*)(&sA[buf][r * LDS_STRIDE + c]) = pk;
      }
    }
#pragma unroll
    for (int i = 0; i < 4; ++i) {
      int idx = tid + i * 256;
      int n = idx & 127;
      int k0 = (idx >> 7) * 4;
      uint2 pk;
      pk.x = pack2bf(bReg[i * 4 + 0], bReg[i * 4 + 1]);
      pk.y = pack2bf(bReg[i * 4 + 2], bReg[i * 4 + 3]);
      *(uint2*)(&sB[buf][n * LDS_STRIDE + k0]) = pk;   // transposed [n][k]
    }
  };

  v8f acc[4][2];
  const v8f vzero = {0.f, 0.f, 0.f, 0.f, 0.f, 0.f, 0.f, 0.f};
#pragma unroll
  for (int i = 0; i < 4; ++i)
#pragma unroll
    for (int j = 0; j < 2; ++j) acc[i][j] = vzero;

  const int nk = Kd / BK;

  // prologue: fill buffer 0
  issueA(0);
  issueB(0);
  stage(0);
  __syncthreads();

  for (int kt = 0; kt < nk; ++kt) {
    const int cur = kt & 1;

    // issue next tile's global loads early; the loadcnt wait lands in stage()
    // *after* the WMMA block below.
    if (kt + 1 < nk) {
      issueA(kt + 1);
      issueB(kt + 1);
      if (kt + 2 < nk)  // hint the tile after next into L2
        __builtin_prefetch(Bp + (size_t)((kt + 2) * BK + (tid >> 5)) * ldb +
                               colBase + (size_t)(tid & 31) * 4, 0, 0);
    }

    v16bf aF[4], bF[2];
#pragma unroll
    for (int mt = 0; mt < 4; ++mt)
      aF[mt] = load_fragA(&sA[cur][(mBase + mt * 16 + ln16) * LDS_STRIDE], half);
#pragma unroll
    for (int nt = 0; nt < 2; ++nt)
      bF[nt] = load_fragB(&sB[cur][(nBase + nt * 16 + ln16) * LDS_STRIDE], half);

#pragma unroll
    for (int mt = 0; mt < 4; ++mt)
#pragma unroll
      for (int nt = 0; nt < 2; ++nt)
        acc[mt][nt] = __builtin_amdgcn_wmma_f32_16x16x32_bf16(
            false, aF[mt], false, bF[nt], (short)0, acc[mt][nt], false, false);

    if (kt + 1 < nk) stage(cur ^ 1);
    __syncthreads();
  }

  // ---- epilogue: C/D layout -> row = r + 8*half, col = lane%16 ----
#pragma unroll
  for (int mt = 0; mt < 4; ++mt) {
#pragma unroll
    for (int nt = 0; nt < 2; ++nt) {
#pragma unroll
      for (int r = 0; r < 8; ++r) {
        size_t row = rowBase + mBase + mt * 16 + r + half * 8;
        size_t col = colBase + nBase + nt * 16 + ln16;
        float v = acc[mt][nt][r];
        if (OUT_BF16) ((unsigned short*)Cp)[row * (size_t)ldc + col] = f2bf(v);
        else          ((float*)Cp)[row * (size_t)ldc + col] = v;
      }
    }
  }
}

// Causal depthwise conv (K=4) + bias + SiLU, gated by SiLU(z). One thread per
// (b, l, e). Reads bf16 xz = [xc | z], writes bf16 activations.
__global__ __launch_bounds__(256)
void conv_silu_gate(const unsigned short* __restrict__ xz,
                    const float* __restrict__ wconv,
                    const float* __restrict__ bconv,
                    unsigned short* __restrict__ act) {
  const int idx = blockIdx.x * 256 + threadIdx.x;
  const int e = idx & (E_DIM - 1);
  const int l = (idx >> 11) & (L_SEQ - 1);       // E_DIM = 2^11
  const int b = idx >> 23;                       // / (E_DIM * L_SEQ)
  const size_t base = (size_t)b * L_SEQ * (2 * E_DIM);

  float a = bconv[e];
#pragma unroll
  for (int k = 0; k < 4; ++k) {
    int li = l - 3 + k;
    if (li >= 0)
      a = fmaf(wconv[e * 4 + k], bf2f(xz[base + (size_t)li * (2 * E_DIM) + e]), a);
  }
  float z  = bf2f(xz[base + (size_t)l * (2 * E_DIM) + E_DIM + e]);
  float sa = a / (1.0f + __expf(-a));
  float sz = z / (1.0f + __expf(-z));
  act[idx] = f2bf(sa * sz);
}

extern "C" void kernel_launch(void* const* d_in, const int* in_sizes, int n_in,
                              void* d_out, int out_size, void* d_ws, size_t ws_size,
                              hipStream_t stream) {
  const float* x      = (const float*)d_in[0];   // [4,4096,1024]
  const float* w_in   = (const float*)d_in[1];   // [1024,4096]
  const float* w_conv = (const float*)d_in[2];   // [2048,1,4]
  const float* b_conv = (const float*)d_in[3];   // [2048]
  const float* w_out  = (const float*)d_in[7];   // [2048,1024]
  // d_in[4..6], d_in[8..9] (w_sel, w_dt, b_dt, A, Dp) are dead in the reference.

  unsigned short* xz  = (unsigned short*)d_ws;               // bf16 [16384][4096] (128 MB)
  unsigned short* act = xz + (size_t)16384 * 4096;           // bf16 [16384][2048] ( 64 MB)

  dim3 blk(256);

  // GEMM1: xz = x @ w_in   (M=16384, K=1024, N=4096)
  dim3 g1(4096 / BN, 16384 / BM);
  gemm_bf16_wmma<false, true><<<g1, blk, 0, stream>>>(x, w_in, xz,
                                                      1024, 1024, 4096, 4096);

  // conv + SiLU + gate -> act (bf16)
  dim3 g2((unsigned)((size_t)B_SZ * L_SEQ * E_DIM / 256));
  conv_silu_gate<<<g2, blk, 0, stream>>>(xz, w_conv, b_conv, act);

  // GEMM2: out = act @ w_out  (M=16384, K=2048, N=1024), fp32 output
  dim3 g3(1024 / BN, 16384 / BM);
  gemm_bf16_wmma<true, false><<<g3, blk, 0, stream>>>(act, w_out, (float*)d_out,
                                                      2048, 2048, 1024, 1024);
}
